// EdgeInOutGTModel_7576322310513
// MI455X (gfx1250) — matrix operand order repaired and checked by reading, not compiled, not run on hardware
//
#include <hip/hip_runtime.h>
#include <math.h>

// ---------------- model constants ----------------
#define HID   128
#define HEADS 8
#define HD    16
#define LAYERS 3
#define MULT  4
#define NB    8          // batches
#define NN    256        // nodes
#define QW    (3 * HID)  // 384
#define BN    (NB * NN)  // 2048

typedef __attribute__((ext_vector_type(16))) __bf16 v16bf;
typedef __attribute__((ext_vector_type(8)))  __bf16 v8bf;
typedef __attribute__((ext_vector_type(8)))  float  v8f;

__device__ __forceinline__ v8f wmma_bf16(v16bf a, v16bf b, v8f c) {
  return __builtin_amdgcn_wmma_f32_16x16x32_bf16(false, a, false, b, (short)0, c, false, false);
}

// ---------------- weight prep: transpose+bf16 convert, fold edge weights ----------------
__global__ void prep_weights(const float* __restrict__ Wh, const float* __restrict__ Wm1,
                             const float* __restrict__ Wm2, const float* __restrict__ We,
                             const float* __restrict__ Wedge,
                             __bf16* __restrict__ WhT, __bf16* __restrict__ Wm1T,
                             __bf16* __restrict__ Wm2T, float* __restrict__ wfold) {
  long gid = (long)blockIdx.x * 256 + threadIdx.x;
  const long NWh = 3L * 128 * 384, NM1 = 3L * 128 * 512, NM2 = 3L * 512 * 128;
  if (gid < 48) {  // wfold[l*16 + hh] = sum_c W_edge[c] * W_e[l][c][hh]
    int l = (int)(gid >> 4), hh = (int)(gid & 15);
    float s = 0.f;
    for (int c = 0; c < 128; c++) s += Wedge[c] * We[((long)l * 128 + c) * 16 + hh];
    wfold[gid] = s;
  }
  if (gid < NWh) {                       // WhT[l][n=384][k=128]
    long l = gid / (384 * 128), rem = gid - l * 384 * 128;
    long n = rem >> 7, k = rem & 127;
    WhT[gid] = (__bf16)Wh[(l * 128 + k) * 384 + n];
  } else if (gid < NWh + NM1) {          // Wm1T[l][n=512][k=128]
    long t = gid - NWh;
    long l = t / (512 * 128), rem = t - l * 512 * 128;
    long n = rem >> 7, k = rem & 127;
    Wm1T[t] = (__bf16)Wm1[(l * 128 + k) * 512 + n];
  } else if (gid < NWh + NM1 + NM2) {    // Wm2T[l][n=128][k=512]
    long t = gid - NWh - NM1;
    long l = t / (128 * 512), rem = t - l * 128 * 512;
    long n = rem / 512, k = rem - n * 512;
    Wm2T[t] = (__bf16)Wm2[(l * 512 + k) * 128 + n];
  }
}

// ---------------- h0 = node @ W_node (3 -> 128) ----------------
__global__ void in_proj(const float* __restrict__ node, const float* __restrict__ Wn,
                        float* __restrict__ h) {
  int gid = blockIdx.x * 256 + threadIdx.x;   // over BN*128
  int c = gid & 127, rn = gid >> 7;
  const float* nf = node + (size_t)rn * 3;
  h[gid] = nf[0] * Wn[c] + nf[1] * Wn[128 + c] + nf[2] * Wn[256 + c];
}

// ---------------- LayerNorm (wave per row), optional residual add, bf16 out ----------------
__global__ void ln_kernel(const float* __restrict__ X, const float* __restrict__ Xadd,
                          const float* __restrict__ g, const float* __restrict__ b,
                          __bf16* __restrict__ out) {
  int wv = threadIdx.x >> 5, lane = threadIdx.x & 31;
  int row = blockIdx.x * 8 + wv;
  const float* xr = X + (size_t)row * HID;
  float4 xv = *(const float4*)(xr + lane * 4);
  float v0[4] = {xv.x, xv.y, xv.z, xv.w};
  if (Xadd) {
    float4 av = *(const float4*)(Xadd + (size_t)row * HID + lane * 4);
    v0[0] += av.x; v0[1] += av.y; v0[2] += av.z; v0[3] += av.w;
  }
  float s = v0[0] + v0[1] + v0[2] + v0[3];
  float sq = v0[0] * v0[0] + v0[1] * v0[1] + v0[2] * v0[2] + v0[3] * v0[3];
  for (int o = 16; o; o >>= 1) { s += __shfl_xor(s, o, 32); sq += __shfl_xor(sq, o, 32); }
  float mean = s * (1.0f / HID);
  float var  = sq * (1.0f / HID) - mean * mean;
  float inv  = rsqrtf(var + 1e-5f);
  #pragma unroll
  for (int t = 0; t < 4; t++) {
    int c = lane * 4 + t;
    out[(size_t)row * HID + c] = (__bf16)((v0[t] - mean) * inv * g[c] + b[c]);
  }
}

// ---------------- generic wave-tile bf16 WMMA GEMM: C = A(MxK) @ BT(NcxK)^T ----------------
// EPI: 0 = store bf16, 1 = relu->bf16, 2 = f32 + residual
template <int EPI>
__global__ void gemm_kernel(const __bf16* __restrict__ A, const __bf16* __restrict__ BT,
                            __bf16* __restrict__ Cb, float* __restrict__ Cf,
                            const float* __restrict__ Yres, int K, int Nc) {
  int wv = threadIdx.x >> 5, lane = threadIdx.x & 31;
  int lm = lane & 15, lh = lane >> 4;
  int m0 = blockIdx.x * 16;
  int n0 = blockIdx.y * 128 + wv * 16;
  const __bf16* arow = A + (size_t)(m0 + lm) * K + lh * 8;
  const __bf16* brow = BT + (size_t)(n0 + lm) * K + lh * 8;
  v8f acc = {};
  for (int k0 = 0; k0 < K; k0 += 32) {
    union { v16bf v; v8bf h[2]; } a, bm;
    a.h[0] = *(const v8bf*)(arow + k0);        // K = base+0..7
    a.h[1] = *(const v8bf*)(arow + k0 + 16);   // K = base+16..23
    bm.h[0] = *(const v8bf*)(brow + k0);       // K = base+0..7
    bm.h[1] = *(const v8bf*)(brow + k0 + 16);  // K = base+16..23
    acc = wmma_bf16(a.v, bm.v, acc);
  }
  #pragma unroll
  for (int r = 0; r < 8; r++) {
    size_t idx = (size_t)(m0 + r + lh * 8) * Nc + n0 + lm;
    float v = acc[r];
    if (EPI == 0)      Cb[idx] = (__bf16)v;
    else if (EPI == 1) Cb[idx] = (__bf16)fmaxf(v, 0.f);
    else               Cf[idx] = v + Yres[idx];
  }
}

// ---------------- fused attention: per (16-row tile, head, batch) ----------------
__global__ void attn_kernel(const __bf16* __restrict__ qkv, const float* __restrict__ edge,
                            const float* __restrict__ wfold, float* __restrict__ Y) {
  int itile = blockIdx.x, head = blockIdx.y, bId = blockIdx.z;
  int tid = threadIdx.x, wv = tid >> 5, lane = tid & 31;
  int lm = lane & 15, lh = lane >> 4;
  int i0 = itile * 16;
  float wf1 = wfold[head], wf2 = wfold[8 + head];
  __shared__ float sc[16][NN];                   // 16 KB raw scores
  __shared__ __align__(32) __bf16 attb[16][NN];  // 8 KB softmaxed*e2
  __shared__ float ytile[16][16];
  const __bf16* qb = qkv + (size_t)bId * NN * QW;
  const float*  eb = edge + (size_t)bId * NN * NN;

  // phase 1: 16 waves, wave wv -> score tile j0 = wv*16, via WMMA (K=16 padded to 32)
  {
    int j0 = wv * 16;
    union { v16bf v; v8bf h[2]; } a;
    a.h[0] = *(const v8bf*)(qb + (size_t)(i0 + lm) * QW + head * 16 + lh * 8); // q, d=base..base+7
    #pragma unroll
    for (int e = 0; e < 8; e++) a.v[8 + e] = (__bf16)0.0f;                     // pad K 16..31
    v16bf bm;
    #pragma unroll
    for (int e = 0; e < 16; e++) bm[e] = (__bf16)0.0f;
    #pragma unroll
    for (int e = 0; e < 8; e++)  // elements 0..7 carry d=lh*8..lh*8+7; elements 8..15 are the zero pad
      bm[e] = qb[(size_t)(j0 + lm) * QW + HID + head * 16 + lh * 8 + e];
    v8f acc = {};
    acc = wmma_bf16(a.v, bm, acc);
    #pragma unroll
    for (int r = 0; r < 8; r++) {
      int mi = r + lh * 8;
      int j  = j0 + lm;
      sc[mi][j] = acc[r] * 0.25f + eb[(size_t)(i0 + mi) * NN + j] * wf1;
    }
  }
  if (tid < 256) ytile[tid >> 4][tid & 15] = 0.f;
  __syncthreads();

  // phase 2: softmax row wv (wave per row), fuse * (edge*wf2), write bf16
  {
    int r = wv;
    float vals[8];
    float mx = -3.0e38f;
    #pragma unroll
    for (int t = 0; t < 8; t++) { vals[t] = sc[r][lane + 32 * t]; mx = fmaxf(mx, vals[t]); }
    for (int o = 16; o; o >>= 1) mx = fmaxf(mx, __shfl_xor(mx, o, 32));
    float sum = 0.f;
    #pragma unroll
    for (int t = 0; t < 8; t++) { vals[t] = __expf(vals[t] - mx); sum += vals[t]; }
    for (int o = 16; o; o >>= 1) sum += __shfl_xor(sum, o, 32);
    float inv = 1.0f / sum;
    const float* er = eb + (size_t)(i0 + r) * NN;
    #pragma unroll
    for (int t = 0; t < 8; t++) {
      int j = lane + 32 * t;
      attb[r][j] = (__bf16)(vals[t] * inv * (er[j] * wf2));
    }
  }
  __syncthreads();

  // phase 3: y = att @ v ; 8 waves split K=256 (wave wv -> k-chunk wv*32), LDS f32 reduce
  if (wv < 8) {
    int k0 = wv * 32;
    union { v16bf v; v8bf h[2]; } a;
    a.h[0] = *(const v8bf*)(&attb[lm][k0 + lh * 8]);
    a.h[1] = *(const v8bf*)(&attb[lm][k0 + lh * 8 + 16]);
    v16bf bm;
    #pragma unroll
    for (int e = 0; e < 16; e++)
      bm[e] = qb[(size_t)(k0 + lh * 8 + (e & 7) + (e >> 3) * 16) * QW + 2 * HID + head * 16 + lm];  // v column
    v8f acc = {};
    acc = wmma_bf16(a.v, bm, acc);
    #pragma unroll
    for (int r = 0; r < 8; r++) atomicAdd(&ytile[r + lh * 8][lm], acc[r]);
  }
  __syncthreads();
  if (tid < 256) {
    int mi = tid >> 4, d = tid & 15;
    Y[((size_t)bId * NN + i0 + mi) * HID + head * 16 + d] = ytile[mi][d];
  }
}

// ---------------- decode: a[b,j]=h.Wd[:128], c[b,i]=h.Wd[128:], out=10*tanh((a+c)/sqrt128) ----
__global__ void dec_rowdot(const float* __restrict__ h, const float* __restrict__ Wd,
                           float* __restrict__ avec, float* __restrict__ cvec) {
  int wv = threadIdx.x >> 5, lane = threadIdx.x & 31;
  int row = blockIdx.x * 8 + wv;
  const float* hr = h + (size_t)row * HID;
  float sa = 0.f, sc_ = 0.f;
  #pragma unroll
  for (int t = 0; t < 4; t++) {
    int c = lane + 32 * t;
    float x = hr[c];
    sa  += x * Wd[c];
    sc_ += x * Wd[128 + c];
  }
  for (int o = 16; o; o >>= 1) { sa += __shfl_xor(sa, o, 32); sc_ += __shfl_xor(sc_, o, 32); }
  if (lane == 0) { avec[row] = sa; cvec[row] = sc_; }
}

__global__ void dec_out(const float* __restrict__ avec, const float* __restrict__ cvec,
                        float* __restrict__ out) {
  int bi = blockIdx.x;          // b*256 + i
  int j  = threadIdx.x;
  int b = bi >> 8, i = bi & 255;
  float x = (avec[b * NN + j] + cvec[b * NN + i]) * 0.08838834764831845f;
  out[(size_t)bi * NN + j] = 10.0f * tanhf(x);
}

// ---------------- host launch ----------------
extern "C" void kernel_launch(void* const* d_in, const int* in_sizes, int n_in,
                              void* d_out, int out_size, void* d_ws, size_t ws_size,
                              hipStream_t stream) {
  const float* node  = (const float*)d_in[0];
  const float* edge  = (const float*)d_in[1];
  const float* Wnode = (const float*)d_in[2];
  const float* Wedge = (const float*)d_in[3];
  const float* Wh    = (const float*)d_in[4];
  const float* We    = (const float*)d_in[5];
  const float* g1    = (const float*)d_in[6];
  const float* b1    = (const float*)d_in[7];
  const float* g2    = (const float*)d_in[8];
  const float* b2    = (const float*)d_in[9];
  const float* Wm1   = (const float*)d_in[10];
  const float* Wm2   = (const float*)d_in[11];
  const float* Wdec  = (const float*)d_in[12];
  float* out = (float*)d_out;

  // workspace carve-up (~7.9 MB), 256B aligned slices
  char* ws = (char*)d_ws;
  size_t off = 0;
  auto take = [&](size_t bytes) { char* p = ws + off; off = (off + bytes + 255) & ~(size_t)255; return p; };
  float*  h_buf = (float*) take((size_t)BN * HID * 4);
  __bf16* hn    = (__bf16*)take((size_t)BN * HID * 2);
  __bf16* qkv   = (__bf16*)take((size_t)BN * QW  * 2);
  float*  y_buf = (float*) take((size_t)BN * HID * 4);
  __bf16* z_buf = (__bf16*)take((size_t)BN * HID * 2);
  __bf16* hid   = (__bf16*)take((size_t)BN * MULT * HID * 2);
  __bf16* WhT   = (__bf16*)take((size_t)3 * 384 * 128 * 2);
  __bf16* Wm1T  = (__bf16*)take((size_t)3 * 512 * 128 * 2);
  __bf16* Wm2T  = (__bf16*)take((size_t)3 * 128 * 512 * 2);
  float*  wfold = (float*) take(48 * 4);
  float*  avec  = (float*) take((size_t)BN * 4);
  float*  cvec  = (float*) take((size_t)BN * 4);
  (void)ws_size; (void)n_in; (void)in_sizes; (void)out_size;

  // 1) weights: transpose + bf16, fold edge weights
  prep_weights<<<2112, 256, 0, stream>>>(Wh, Wm1, Wm2, We, Wedge, WhT, Wm1T, Wm2T, wfold);
  // 2) input projection
  in_proj<<<(BN * HID) / 256, 256, 0, stream>>>(node, Wnode, h_buf);

  for (int l = 0; l < LAYERS; l++) {
    const __bf16* WhTl  = WhT  + (size_t)l * 384 * 128;
    const __bf16* Wm1Tl = Wm1T + (size_t)l * 512 * 128;
    const __bf16* Wm2Tl = Wm2T + (size_t)l * 128 * 512;
    // LN1 -> bf16
    ln_kernel<<<BN / 8, 256, 0, stream>>>(h_buf, nullptr, g1 + l * HID, b1 + l * HID, hn);
    // qkv = hn @ W_h  (2048x128 @ 128x384)
    gemm_kernel<0><<<dim3(BN / 16, QW / 128), 256, 0, stream>>>(hn, WhTl, qkv, nullptr, nullptr, HID, QW);
    // attention (fused scores + edge bias + softmax*e2 + att@v)
    attn_kernel<<<dim3(NN / 16, HEADS, NB), 512, 0, stream>>>(qkv, edge, wfold + l * 16, y_buf);
    // LN2 of (y + h) -> bf16
    ln_kernel<<<BN / 8, 256, 0, stream>>>(y_buf, h_buf, g2 + l * HID, b2 + l * HID, z_buf);
    // hid = relu(z @ W_mlp1)  (2048x128 @ 128x512)
    gemm_kernel<1><<<dim3(BN / 16, (MULT * HID) / 128), 256, 0, stream>>>(z_buf, Wm1Tl, hid, nullptr, nullptr, HID, MULT * HID);
    // h = hid @ W_mlp2 + y   (2048x512 @ 512x128)
    gemm_kernel<2><<<dim3(BN / 16, 1), 256, 0, stream>>>(hid, Wm2Tl, nullptr, h_buf, y_buf, MULT * HID, HID);
  }

  // decode
  dec_rowdot<<<BN / 8, 256, 0, stream>>>(h_buf, Wdec, avec, cvec);
  dec_out<<<BN, 256, 0, stream>>>(avec, cvec, out);
}
